// GlobalAggregator_55980603736507
// MI455X (gfx1250) — matrix operand order repaired and checked by reading, not compiled, not run on hardware
//
#include <hip/hip_runtime.h>
#include <hip/hip_bf16.h>
#include <math.h>

// ---------------- CDNA5 WMMA types ----------------
typedef _Float16 f16_t;
typedef __attribute__((ext_vector_type(16))) _Float16 v16h;
typedef __attribute__((ext_vector_type(8)))  _Float16 v8h;
typedef __attribute__((ext_vector_type(8)))  float    v8f;
typedef __attribute__((ext_vector_type(2)))  float    v2f;

// ---------------- problem dims ----------------
#define B_SZ   2
#define S_LEN  128
#define L_LEN  64
#define D_MOD  768
#define D_INN  1536
#define N_ST   64
#define D_CONV 4

// GEMM tile: 128x64 block, 8 waves, each wave a 32x32 quadrant (2x2 WMMA tiles)
#define TBM 128
#define TBN 64
#define TBK 32
#define LDSP 40   // padded LDS stride in halves (20 dwords) -> conflict-free frags

__device__ inline float silu_f(float x) { return x / (1.0f + expf(-x)); }
__device__ inline float sigm_f(float x) { return 1.0f / (1.0f + expf(-x)); }

// =====================================================================
// Generic batched WMMA GEMM:  C[bz] = act( alpha * A[bz]@B[bz] + bias[bz] )
// A: MxK row-major (lda); B: KxN (ldb) or transB (NxK); C: MxN (ldc).
// Software-pipelined: next K-tile is prefetched from global into registers
// while the current tile's fragments and WMMAs issue. B staged col-major
// in LDS so all fragment gathers are ds_load_b128 pairs.
// =====================================================================
__global__ __launch_bounds__(256) void gemm_wmma_kernel(
    const float* __restrict__ A, int lda, long long strideA,
    const float* __restrict__ Bw, int ldb, long long strideB, int transB,
    const float* __restrict__ bias, long long strideBias,
    float* __restrict__ C, int ldc, long long strideC,
    int M, int N, int K, float alpha, int act)
{
  __shared__ f16_t As[TBM * LDSP];   // row-major rows x K   (10 KB)
  __shared__ f16_t Bs[TBN * LDSP];   // col-major cols x K   (5 KB)

  const int tid = threadIdx.x;
  const int bz  = blockIdx.z;
  const float* Ab = A + (size_t)bz * strideA;
  const float* Bb = Bw + (size_t)bz * strideB;
  float*       Cb = C + (size_t)bz * strideC;
  const float* biasp = bias ? (bias + (size_t)bz * strideBias) : nullptr;

  const int blockM = blockIdx.y * TBM;
  const int blockN = blockIdx.x * TBN;
  const int wave = tid >> 5;
  const int lane = tid & 31;
  const int wm = (wave & 3) * 32;    // wave's 32-row group
  const int wn = (wave >> 2) * 32;   // wave's 32-col group
  const int lidx = lane & 15;
  const int lhi  = lane >> 4;

  const bool interior = (blockM + TBM <= M) && (blockN + TBN <= N) &&
                        ((K & (TBK - 1)) == 0);
  // staging coordinates: A 128x32 (16 elems/thread), B 32x64 (8 elems/thread)
  const int arow = tid >> 1;          // 0..127
  const int acol = (tid & 1) * 16;    // 0 or 16
  const int bk   = tid >> 3;          // 0..31
  const int bc   = (tid & 7) * 8;     // 0..56

  float aR[16]; float bR[8];

  auto gload = [&](int k0) {
    if (interior) {
      const float* ag = Ab + (size_t)(blockM + arow) * lda + k0 + acol;
      float4 q0 = *(const float4*)(ag);
      float4 q1 = *(const float4*)(ag + 4);
      float4 q2 = *(const float4*)(ag + 8);
      float4 q3 = *(const float4*)(ag + 12);
      aR[0] = q0.x; aR[1] = q0.y; aR[2]  = q0.z; aR[3]  = q0.w;
      aR[4] = q1.x; aR[5] = q1.y; aR[6]  = q1.z; aR[7]  = q1.w;
      aR[8] = q2.x; aR[9] = q2.y; aR[10] = q2.z; aR[11] = q2.w;
      aR[12] = q3.x; aR[13] = q3.y; aR[14] = q3.z; aR[15] = q3.w;
      if (!transB) {
        const float* bg = Bb + (size_t)(k0 + bk) * ldb + blockN + bc;
        float4 b0v = *(const float4*)bg;
        float4 b1v = *(const float4*)(bg + 4);
        bR[0] = b0v.x; bR[1] = b0v.y; bR[2] = b0v.z; bR[3] = b0v.w;
        bR[4] = b1v.x; bR[5] = b1v.y; bR[6] = b1v.z; bR[7] = b1v.w;
      } else {
#pragma unroll
        for (int j = 0; j < 8; ++j)
          bR[j] = Bb[(size_t)(blockN + bc + j) * ldb + k0 + bk];
      }
    } else {
      const int grow = blockM + arow;
#pragma unroll
      for (int j = 0; j < 16; ++j) {
        int gk = k0 + acol + j;
        aR[j] = (grow < M && gk < K) ? Ab[(size_t)grow * lda + gk] : 0.0f;
      }
      const int gk2 = k0 + bk;
#pragma unroll
      for (int j = 0; j < 8; ++j) {
        int gcol = blockN + bc + j;
        bR[j] = (gcol < N && gk2 < K)
                    ? (transB ? Bb[(size_t)gcol * ldb + gk2]
                              : Bb[(size_t)gk2 * ldb + gcol])
                    : 0.0f;
      }
    }
  };

  auto lstore = [&]() {
    v8h lo, hi;
#pragma unroll
    for (int j = 0; j < 8; ++j) { lo[j] = (f16_t)aR[j]; hi[j] = (f16_t)aR[j + 8]; }
    *(v8h*)&As[arow * LDSP + acol]     = lo;
    *(v8h*)&As[arow * LDSP + acol + 8] = hi;
#pragma unroll
    for (int j = 0; j < 8; ++j)
      Bs[(bc + j) * LDSP + bk] = (f16_t)bR[j];
  };

  v8f c00 = {}, c01 = {}, c10 = {}, c11 = {};

  gload(0);
  for (int k0 = 0; k0 < K; k0 += TBK) {
    lstore();
    __syncthreads();
    if (k0 + TBK < K) gload(k0 + TBK);   // prefetch next tile during compute

    // fragments: two ds_load_b128 per operand
    const v8h* ap0 = (const v8h*)&As[(wm + lidx) * LDSP + lhi * 8];
    v16h a0 = __builtin_shufflevector(ap0[0], ap0[2], 0, 1, 2, 3, 4, 5, 6, 7,
                                      8, 9, 10, 11, 12, 13, 14, 15);
    const v8h* ap1 = (const v8h*)&As[(wm + 16 + lidx) * LDSP + lhi * 8];
    v16h a1 = __builtin_shufflevector(ap1[0], ap1[2], 0, 1, 2, 3, 4, 5, 6, 7,
                                      8, 9, 10, 11, 12, 13, 14, 15);
    const v8h* bp0 = (const v8h*)&Bs[(wn + lidx) * LDSP + lhi * 8];
    v16h b0 = __builtin_shufflevector(bp0[0], bp0[2], 0, 1, 2, 3, 4, 5, 6, 7,
                                      8, 9, 10, 11, 12, 13, 14, 15);
    const v8h* bp1 = (const v8h*)&Bs[(wn + 16 + lidx) * LDSP + lhi * 8];
    v16h b1 = __builtin_shufflevector(bp1[0], bp1[2], 0, 1, 2, 3, 4, 5, 6, 7,
                                      8, 9, 10, 11, 12, 13, 14, 15);

    c00 = __builtin_amdgcn_wmma_f32_16x16x32_f16(false, a0, false, b0,
                                                 (short)0, c00, false, false);
    c01 = __builtin_amdgcn_wmma_f32_16x16x32_f16(false, a0, false, b1,
                                                 (short)0, c01, false, false);
    c10 = __builtin_amdgcn_wmma_f32_16x16x32_f16(false, a1, false, b0,
                                                 (short)0, c10, false, false);
    c11 = __builtin_amdgcn_wmma_f32_16x16x32_f16(false, a1, false, b1,
                                                 (short)0, c11, false, false);
    __syncthreads();
  }

  // Epilogue: lane holds col (lane&15), rows (lane>>4)*8 + i
  const int col0 = blockN + wn + lidx;
  const int col1 = col0 + 16;
  const int rowT0 = blockM + wm + lhi * 8;
  const int rowT1 = rowT0 + 16;
  const float bv0 = (biasp && col0 < N) ? biasp[col0] : 0.0f;
  const float bv1 = (biasp && col1 < N) ? biasp[col1] : 0.0f;

  auto cstore = [&](const v8f& cc, int rbase, int col, float bval) {
    if (col >= N) return;
#pragma unroll
    for (int i = 0; i < 8; ++i) {
      int row = rbase + i;
      if (row < M) {
        float v = cc[i] * alpha + bval;
        if (act == 1) v = tanhf(v);
        Cb[(size_t)row * ldc + col] = v;
      }
    }
  };
  cstore(c00, rowT0, col0, bv0);
  cstore(c01, rowT0, col1, bv1);
  cstore(c10, rowT1, col0, bv0);
  cstore(c11, rowT1, col1, bv1);
}

// =====================================================================
// Scene pooling
// =====================================================================
__global__ __launch_bounds__(256) void pool_kernel(
    const float* __restrict__ scene, const float* __restrict__ hdn,
    const float* __restrict__ sp_w2, const float* __restrict__ sp_b2,
    float* __restrict__ movie)
{
  __shared__ float wls[L_LEN];
  int bs = blockIdx.x;
  int tid = threadIdx.x;
  if (tid < L_LEN) {
    const float* hr = hdn + ((size_t)bs * L_LEN + tid) * 384;
    float acc = sp_b2[0];
    for (int k = 0; k < 384; ++k) acc += hr[k] * sp_w2[k];
    wls[tid] = acc;
  }
  __syncthreads();
  if (tid == 0) {
    float m = -1e30f;
    for (int l = 0; l < L_LEN; ++l) m = fmaxf(m, wls[l]);
    float s = 0.0f;
    for (int l = 0; l < L_LEN; ++l) { wls[l] = expf(wls[l] - m); s += wls[l]; }
    float inv = 1.0f / s;
    for (int l = 0; l < L_LEN; ++l) wls[l] *= inv;
  }
  __syncthreads();
  const float* sr = scene + (size_t)bs * L_LEN * D_MOD;
  for (int d = tid; d < D_MOD; d += 256) {
    float acc = 0.0f;
    for (int l = 0; l < L_LEN; ++l) acc += wls[l] * sr[(size_t)l * D_MOD + d];
    movie[(size_t)bs * D_MOD + d] = acc;
  }
}

// =====================================================================
// Adjacency mask + row softmax (4 batches = head*b)
// =====================================================================
__global__ __launch_bounds__(128) void mask_softmax_kernel(
    float* __restrict__ scores, const float* __restrict__ causal,
    const float* __restrict__ charadj)
{
  __shared__ float red[S_LEN];
  int i = blockIdx.x;
  int z = blockIdx.y;
  int h = z >> 1, b = z & 1;
  int j = threadIdx.x;
  const float* adj = (h == 0 ? causal : charadj) + ((size_t)b * S_LEN + i) * S_LEN;
  float* row = scores + ((size_t)z * S_LEN + i) * S_LEN;
  float a = adj[j];
  float sc = row[j] * (a + 1e-6f) + (a == 0.0f ? -10000.0f : 0.0f);
  red[j] = sc; __syncthreads();
  for (int off = 64; off > 0; off >>= 1) {
    if (j < off) red[j] = fmaxf(red[j], red[j + off]);
    __syncthreads();
  }
  float mx = red[0]; __syncthreads();
  float e = expf(sc - mx);
  red[j] = e; __syncthreads();
  for (int off = 64; off > 0; off >>= 1) {
    if (j < off) red[j] += red[j + off];
    __syncthreads();
  }
  row[j] = e / red[0];
}

__device__ inline float blk_reduce_256(float v, float* red) {
  int tid = threadIdx.x;
  red[tid] = v; __syncthreads();
  for (int off = 128; off > 0; off >>= 1) {
    if (tid < off) red[tid] += red[tid + off];
    __syncthreads();
  }
  float r = red[0]; __syncthreads();
  return r;
}

// =====================================================================
// hgt_out = LN(movie + fused); normed = RMS(hgt_out)
// =====================================================================
__global__ __launch_bounds__(256) void add_ln_rms_kernel(
    const float* __restrict__ movie, const float* __restrict__ fused,
    const float* __restrict__ g, const float* __restrict__ bta,
    const float* __restrict__ rmsw, float* __restrict__ normed)
{
  __shared__ float red[256];
  int row = blockIdx.x, tid = threadIdx.x;
  float loc[3]; float s = 0.0f;
#pragma unroll
  for (int j = 0; j < 3; ++j) {
    int c = tid * 3 + j;
    loc[j] = movie[(size_t)row * D_MOD + c] + fused[(size_t)row * D_MOD + c];
    s += loc[j];
  }
  float mean = blk_reduce_256(s, red) * (1.0f / D_MOD);
  float v = 0.0f;
#pragma unroll
  for (int j = 0; j < 3; ++j) { float dv = loc[j] - mean; v += dv * dv; }
  float var = blk_reduce_256(v, red) * (1.0f / D_MOD);
  float inv = rsqrtf(var + 1e-5f);
  float ss = 0.0f;
#pragma unroll
  for (int j = 0; j < 3; ++j) {
    int c = tid * 3 + j;
    float y = (loc[j] - mean) * inv * g[c] + bta[c];
    loc[j] = y; ss += y * y;
  }
  float ms = blk_reduce_256(ss, red) * (1.0f / D_MOD);
  float rinv = rsqrtf(ms + 1e-5f);
#pragma unroll
  for (int j = 0; j < 3; ++j) {
    int c = tid * 3 + j;
    normed[(size_t)row * D_MOD + c] = loc[j] * rinv * rmsw[c];
  }
}

// Mamba-internal layernorm with optional sequence flip
__global__ __launch_bounds__(256) void ln_flip_kernel(
    const float* __restrict__ x, const float* __restrict__ g,
    const float* __restrict__ bta, float* __restrict__ out, int flip)
{
  __shared__ float red[256];
  int row = blockIdx.x, tid = threadIdx.x;
  int bb = row / S_LEN, sIdx = row % S_LEN;
  int srow = flip ? (bb * S_LEN + (S_LEN - 1 - sIdx)) : row;
  const float* xr = x + (size_t)srow * D_MOD;
  float loc[3]; float s = 0.0f;
#pragma unroll
  for (int j = 0; j < 3; ++j) { int c = tid * 3 + j; loc[j] = xr[c]; s += loc[j]; }
  float mean = blk_reduce_256(s, red) * (1.0f / D_MOD);
  float v = 0.0f;
#pragma unroll
  for (int j = 0; j < 3; ++j) { float dv = loc[j] - mean; v += dv * dv; }
  float var = blk_reduce_256(v, red) * (1.0f / D_MOD);
  float inv = rsqrtf(var + 1e-5f);
#pragma unroll
  for (int j = 0; j < 3; ++j) {
    int c = tid * 3 + j;
    out[(size_t)row * D_MOD + c] = (loc[j] - mean) * inv * g[c] + bta[c];
  }
}

// ---------------- small elementwise kernels ----------------
__global__ void xpad_kernel(const float* __restrict__ xz, float* __restrict__ xpad) {
  long long n = (long long)B_SZ * (S_LEN + 3) * D_INN;
  for (long long idx = blockIdx.x * 256LL + threadIdx.x; idx < n;
       idx += (long long)gridDim.x * 256LL) {
    int i = (int)(idx % D_INN);
    long long rt = idx / D_INN;
    int t = (int)(rt % (S_LEN + 3));
    int b = (int)(rt / (S_LEN + 3));
    float v = 0.0f;
    if (t >= 3) v = xz[((size_t)(b * S_LEN + (t - 3))) * (2 * D_INN) + i];
    xpad[idx] = v;
  }
}

__global__ void wconv_kernel(const float* __restrict__ conv_w, float* __restrict__ wconv) {
  long long n = (long long)D_CONV * D_INN * D_INN;
  for (long long idx = blockIdx.x * 256LL + threadIdx.x; idx < n;
       idx += (long long)gridDim.x * 256LL) {
    int o = (int)(idx % D_INN);
    long long row = idx / D_INN;
    int i = (int)(row % D_INN);
    int k = (int)(row / D_INN);
    wconv[idx] = conv_w[((size_t)o * D_INN + i) * D_CONV + k];
  }
}

__global__ void bias_silu_kernel(const float* __restrict__ xc,
                                 const float* __restrict__ conv_b,
                                 float* __restrict__ xact) {
  long long n = (long long)B_SZ * S_LEN * D_INN;
  for (long long idx = blockIdx.x * 256LL + threadIdx.x; idx < n;
       idx += (long long)gridDim.x * 256LL) {
    int i = (int)(idx % D_INN);
    xact[idx] = silu_f(xc[idx] + conv_b[i]);
  }
}

__global__ void delta_kernel(const float* __restrict__ xssm,
                             const float* __restrict__ dt_w,
                             const float* __restrict__ dt_b,
                             float* __restrict__ delta) {
  long long n = (long long)B_SZ * S_LEN * D_INN;
  for (long long idx = blockIdx.x * 256LL + threadIdx.x; idx < n;
       idx += (long long)gridDim.x * 256LL) {
    long long row = idx / D_INN;
    int d = (int)(idx % D_INN);
    float v = xssm[(size_t)row * 129] * dt_w[d] + dt_b[d];
    delta[idx] = (v > 20.0f) ? v : log1pf(expf(v));
  }
}

__global__ void ygate_kernel(float* __restrict__ y, const float* __restrict__ xz) {
  long long n = (long long)B_SZ * S_LEN * D_INN;
  for (long long idx = blockIdx.x * 256LL + threadIdx.x; idx < n;
       idx += (long long)gridDim.x * 256LL) {
    long long row = idx / D_INN;
    int d = (int)(idx % D_INN);
    y[idx] *= silu_f(xz[(size_t)row * (2 * D_INN) + D_INN + d]);
  }
}

__global__ void combine_kernel(const float* __restrict__ normed,
                               const float* __restrict__ proj0,
                               const float* __restrict__ proj1,
                               float* __restrict__ glob) {
  long long n = (long long)B_SZ * S_LEN * D_MOD;
  for (long long idx = blockIdx.x * 256LL + threadIdx.x; idx < n;
       idx += (long long)gridDim.x * 256LL) {
    int dd = (int)(idx % D_MOD);
    long long r = idx / D_MOD;
    int sIdx = (int)(r % S_LEN);
    int b = (int)(r / S_LEN);
    glob[idx] = 2.0f * normed[idx] + proj0[idx] +
                proj1[((size_t)(b * S_LEN + (S_LEN - 1 - sIdx))) * D_MOD + dd];
  }
}

__global__ void final_gate_kernel(const float* __restrict__ scene,
                                  const float* __restrict__ sg,
                                  const float* __restrict__ gg,
                                  const float* __restrict__ glob,
                                  float* __restrict__ out) {
  long long n = (long long)B_SZ * S_LEN * L_LEN * D_MOD;
  for (long long idx = blockIdx.x * 256LL + threadIdx.x; idx < n;
       idx += (long long)gridDim.x * 256LL) {
    int d = (int)(idx % D_MOD);
    long long row = idx / D_MOD;
    long long bs = row / L_LEN;
    float gate = sigm_f(sg[idx] + gg[(size_t)bs * D_MOD + d]);
    float sc = scene[idx];
    float gl = glob[(size_t)bs * D_MOD + d];
    out[idx] = gate * sc + (1.0f - gate) * gl;
  }
}

// =====================================================================
// Mamba scan: recurrence independent across channels; each block owns a
// 128-channel slice for one batch and runs the full t-loop. The routing
// GEMM (g_ctx @ route_w) runs on the native f32 WMMA (16x16x4) per wave,
// staging route_w in LDS once. hist slice streams through L2 (hist: 96 MiB).
// =====================================================================
__global__ __launch_bounds__(256) void scan_kernel(
    const float* __restrict__ x_act, const float* __restrict__ delta,
    const float* __restrict__ xssm, const float* __restrict__ causal,
    const float* __restrict__ charadj, const float* __restrict__ route_w,
    const float* __restrict__ route_b, float* __restrict__ hist,
    float* __restrict__ y, int dir)
{
  __shared__ float wls[S_LEN];
  __shared__ float sB[N_ST];
  __shared__ float sC[N_ST];
  __shared__ float sRB[N_ST];
  __shared__ float sRW[N_ST * N_ST];      // 16 KB
  __shared__ float gctx[128 * N_ST];      // 32 KB
  __shared__ float sCorr[128 * N_ST];     // 32 KB
  __shared__ float ypart[256];
  __shared__ float sdeg, sany;

  int b = blockIdx.y;
  int tid = threadIdx.x;
  int dloc = tid >> 1;
  int nh = tid & 1;
  int n0 = nh * 32;
  int d = blockIdx.x * 128 + dloc;
  int wave = tid >> 5, lane = tid & 31;
  int lidx = lane & 15, lhi = lane >> 4;
  int rowbase = wave * 16;                // wave's 16 channel rows for WMMA

  for (int i = tid; i < N_ST * N_ST; i += 256) sRW[i] = route_w[i];
  if (tid < N_ST) sRB[tid] = route_b[tid];

  float h[32];
#pragma unroll
  for (int j = 0; j < 32; ++j) h[j] = 0.0f;
  __syncthreads();

  for (int t = 0; t < S_LEN; ++t) {
    if (tid < 128) {
      int s = tid; float wv = 0.0f;
      if (s < t) {
        int tt = dir ? (S_LEN - 1 - t) : t;
        int ssrc = dir ? (S_LEN - 1 - s) : s;
        size_t ai = ((size_t)b * S_LEN + tt) * S_LEN + ssrc;
        wv = 0.5f * (causal[ai] + charadj[ai]);
      }
      wls[s] = wv;
    } else if (tid < 192) {
      sB[tid - 128] = xssm[((size_t)b * S_LEN + t) * 129 + 1 + (tid - 128)];
    } else {
      sC[tid - 192] = xssm[((size_t)b * S_LEN + t) * 129 + 65 + (tid - 192)];
    }
    __syncthreads();
    if (tid == 0) {
      float s = 0.0f, any = 0.0f;
      for (int i = 0; i < S_LEN; ++i) { s += wls[i]; if (wls[i] != 0.0f) any = 1.0f; }
      sdeg = fmaxf(s, 1.0f); sany = any;
    }
    __syncthreads();

    size_t rowoff = ((size_t)b * S_LEN + t) * D_INN + d;
    float xv = x_act[rowoff];
    float ed = expf(-delta[rowoff]);

    float g[32];
#pragma unroll
    for (int j = 0; j < 32; ++j) g[j] = 0.0f;
    for (int s = 0; s < t; ++s) {
      float wv = wls[s];
      const float4* hb = (const float4*)(hist +
          (((size_t)s * B_SZ) + b) * (size_t)D_INN * N_ST + (size_t)d * N_ST + n0);
      if (s + 1 < t)
        __builtin_prefetch(hist + (((size_t)(s + 1) * B_SZ) + b) * (size_t)D_INN * N_ST +
                           (size_t)d * N_ST + n0, 0, 1);
      if (wv != 0.0f) {
#pragma unroll
        for (int q = 0; q < 8; ++q) {
          float4 hv = hb[q];
          g[4 * q + 0] += wv * hv.x;
          g[4 * q + 1] += wv * hv.y;
          g[4 * q + 2] += wv * hv.z;
          g[4 * q + 3] += wv * hv.w;
        }
      }
    }
    float invdeg = 1.0f / sdeg;
#pragma unroll
    for (int j = 0; j < 32; ++j) {
      h[j] = h[j] * ed + sB[n0 + j] * xv;
      gctx[dloc * N_ST + n0 + j] = g[j] * invdeg;
    }
    __syncthreads();

    // routing GEMM: sCorr = gctx(128x64) @ route_w(64x64) via f32 WMMA
#if __has_builtin(__builtin_amdgcn_wmma_f32_16x16x4_f32)
    for (int nt = 0; nt < 4; ++nt) {
      v8f cc = {};
#pragma unroll
      for (int k0 = 0; k0 < N_ST; k0 += 4) {
        v2f avv, bvv;
        int ka = k0 + lhi * 2;
        avv[0] = gctx[(rowbase + lidx) * N_ST + ka];
        avv[1] = gctx[(rowbase + lidx) * N_ST + ka + 1];
        bvv[0] = sRW[ka * N_ST + nt * 16 + lidx];
        bvv[1] = sRW[(ka + 1) * N_ST + nt * 16 + lidx];
        cc = __builtin_amdgcn_wmma_f32_16x16x4_f32(false, avv, false, bvv,
                                                   (short)0, cc, false, false);
      }
#pragma unroll
      for (int i = 0; i < 8; ++i)
        sCorr[(rowbase + lhi * 8 + i) * N_ST + nt * 16 + lidx] = cc[i];
    }
#else
    for (int j = 0; j < 32; ++j) {
      int np = n0 + j;
      float acc = 0.0f;
      for (int n = 0; n < N_ST; ++n)
        acc += gctx[dloc * N_ST + n] * sRW[n * N_ST + np];
      sCorr[dloc * N_ST + np] = acc;
    }
#endif
    __syncthreads();

    float anyf = sany;
#pragma unroll
    for (int j = 0; j < 32; ++j) {
      float acc = sCorr[dloc * N_ST + n0 + j] + sRB[n0 + j];
      h[j] += anyf * 0.1f * silu_f(acc);
    }

    float4* ob = (float4*)(hist +
        (((size_t)t * B_SZ) + b) * (size_t)D_INN * N_ST + (size_t)d * N_ST + n0);
    float yp = 0.0f;
#pragma unroll
    for (int q = 0; q < 8; ++q) {
      float4 hv;
      hv.x = h[4 * q + 0]; hv.y = h[4 * q + 1];
      hv.z = h[4 * q + 2]; hv.w = h[4 * q + 3];
      ob[q] = hv;
      yp += hv.x * sC[n0 + 4 * q + 0] + hv.y * sC[n0 + 4 * q + 1] +
            hv.z * sC[n0 + 4 * q + 2] + hv.w * sC[n0 + 4 * q + 3];
    }
    ypart[tid] = yp;
    __syncthreads();
    if (nh == 0) y[rowoff] = ypart[tid] + ypart[tid + 1];
    __syncthreads();
  }
}

// ---------------- host-side helpers ----------------
static void launch_gemm(hipStream_t st, const float* A, int lda, long long sA,
                        const float* Bw, int ldb, long long sB, int transB,
                        const float* bias, long long sBias,
                        float* C, int ldc, long long sC,
                        int M, int N, int K, int batch, float alpha, int act)
{
  dim3 g((N + TBN - 1) / TBN, (M + TBM - 1) / TBM, batch);
  gemm_wmma_kernel<<<g, 256, 0, st>>>(A, lda, sA, Bw, ldb, sB, transB,
                                      bias, sBias, C, ldc, sC, M, N, K, alpha, act);
}

static inline int nblk(long long n) {
  long long b = (n + 255) / 256;
  if (b > 8192) b = 8192;
  return (int)b;
}

enum {
  IN_SCENE = 0, IN_CAUSAL, IN_CHAR,
  P_SP_W1, P_SP_B1, P_SP_W2, P_SP_B2,
  P_HGT_Q, P_HGT_QB, P_HGT_K, P_HGT_KB, P_HGT_V, P_HGT_VB,
  P_HGT_OUT_W, P_HGT_OUT_B, P_HGT_LN_G, P_HGT_LN_B,
  P_RMS_W, P_LN_G, P_LN_B, P_IN_PROJ, P_CONV_W, P_CONV_B,
  P_X_PROJ, P_DT_W, P_DT_B, P_ROUTE_W, P_ROUTE_B, P_MAMBA_OUT,
  P_RG_W, P_RG_B
};

extern "C" void kernel_launch(void* const* d_in, const int* in_sizes, int n_in,
                              void* d_out, int out_size, void* d_ws, size_t ws_size,
                              hipStream_t stream)
{
  (void)in_sizes; (void)n_in; (void)out_size; (void)ws_size;
  const float* P[31];
  for (int i = 0; i < 31; ++i) P[i] = (const float*)d_in[i];
  const float* scene = P[IN_SCENE];
  const float* causal = P[IN_CAUSAL];
  const float* charadj = P[IN_CHAR];
  float* out = (float*)d_out;

  const int ROWS = B_SZ * S_LEN;            // 256
  const int FLATROWS = ROWS * L_LEN;        // 16384

  // ---------------- workspace layout (floats) ----------------
  float* W = (float*)d_ws;
  size_t o = 0;
  auto alloc = [&](size_t n) { float* p = W + o; o += n; return p; };

  const size_t HIST_N = (size_t)S_LEN * B_SZ * D_INN * N_ST;  // 25.2M floats
  float* histb = alloc(HIST_N);
  float* sg    = histb;                       // alias: sg GEMM runs after scan
  float* hdn   = alloc((size_t)FLATROWS * 384);
  float* movie = alloc((size_t)ROWS * D_MOD);
  float* qb    = alloc((size_t)2 * ROWS * D_MOD);
  float* kb    = alloc((size_t)2 * ROWS * D_MOD);
  float* vb    = alloc((size_t)2 * ROWS * D_MOD);
  float* scoresb = alloc((size_t)4 * S_LEN * S_LEN);
  float* attnb = alloc((size_t)ROWS * (2 * D_MOD));
  float* fusedb = alloc((size_t)ROWS * D_MOD);
  float* normedb = alloc((size_t)ROWS * D_MOD);
  float* xlnb  = alloc((size_t)ROWS * D_MOD);
  float* xzb   = alloc((size_t)ROWS * 2 * D_INN);
  float* xpadb = alloc((size_t)B_SZ * (S_LEN + 3) * D_INN);
  float* wconvb = alloc((size_t)D_CONV * D_INN * D_INN);
  float* xcb   = alloc((size_t)ROWS * D_INN);
  float* xactb = alloc((size_t)ROWS * D_INN);
  float* xssmb = alloc((size_t)ROWS * 129);
  float* deltab = alloc((size_t)ROWS * D_INN);
  float* yb    = alloc((size_t)ROWS * D_INN);
  float* projb0 = alloc((size_t)ROWS * D_MOD);
  float* projb1 = alloc((size_t)ROWS * D_MOD);
  float* globb = alloc((size_t)ROWS * D_MOD);
  float* ggb   = alloc((size_t)ROWS * D_MOD);

  // ---------------- 1. scene pooling ----------------
  launch_gemm(stream, scene, D_MOD, 0, P[P_SP_W1], 384, 0, 0,
              P[P_SP_B1], 0, hdn, 384, 0, FLATROWS, 384, D_MOD, 1, 1.0f, /*tanh*/1);
  pool_kernel<<<ROWS, 256, 0, stream>>>(scene, hdn, P[P_SP_W2], P[P_SP_B2], movie);

  // ---------------- 2. HGT attention ----------------
  const long long WQS = (long long)D_MOD * D_MOD;
  launch_gemm(stream, movie, D_MOD, 0, P[P_HGT_Q], D_MOD, WQS, 0,
              P[P_HGT_QB], D_MOD, qb, D_MOD, (long long)ROWS * D_MOD,
              ROWS, D_MOD, D_MOD, 2, 1.0f, 0);
  launch_gemm(stream, movie, D_MOD, 0, P[P_HGT_K], D_MOD, WQS, 0,
              P[P_HGT_KB], D_MOD, kb, D_MOD, (long long)ROWS * D_MOD,
              ROWS, D_MOD, D_MOD, 2, 1.0f, 0);
  launch_gemm(stream, movie, D_MOD, 0, P[P_HGT_V], D_MOD, WQS, 0,
              P[P_HGT_VB], D_MOD, vb, D_MOD, (long long)ROWS * D_MOD,
              ROWS, D_MOD, D_MOD, 2, 1.0f, 0);

  launch_gemm(stream, qb, D_MOD, (long long)S_LEN * D_MOD,
              kb, D_MOD, (long long)S_LEN * D_MOD, /*transB*/1,
              nullptr, 0, scoresb, S_LEN, (long long)S_LEN * S_LEN,
              S_LEN, S_LEN, D_MOD, 4, 1.0f / sqrtf((float)D_MOD), 0);
  mask_softmax_kernel<<<dim3(S_LEN, 4), S_LEN, 0, stream>>>(scoresb, causal, charadj);

  for (int h = 0; h < 2; ++h) {
    launch_gemm(stream, scoresb + (size_t)h * 2 * S_LEN * S_LEN, S_LEN,
                (long long)S_LEN * S_LEN,
                vb + (size_t)h * ROWS * D_MOD, D_MOD, (long long)S_LEN * D_MOD, 0,
                nullptr, 0,
                attnb + h * D_MOD, 2 * D_MOD, (long long)S_LEN * 2 * D_MOD,
                S_LEN, D_MOD, S_LEN, B_SZ, 1.0f, 0);
  }
  launch_gemm(stream, attnb, 2 * D_MOD, 0, P[P_HGT_OUT_W], D_MOD, 0, 0,
              P[P_HGT_OUT_B], 0, fusedb, D_MOD, 0, ROWS, D_MOD, 2 * D_MOD, 1, 1.0f, 0);
  add_ln_rms_kernel<<<ROWS, 256, 0, stream>>>(movie, fusedb, P[P_HGT_LN_G],
                                              P[P_HGT_LN_B], P[P_RMS_W], normedb);

  wconv_kernel<<<nblk((long long)D_CONV * D_INN * D_INN), 256, 0, stream>>>(
      P[P_CONV_W], wconvb);

  // ---------------- 3. bidirectional Mamba ----------------
  float* projs[2] = { projb0, projb1 };
  for (int dir = 0; dir < 2; ++dir) {
    ln_flip_kernel<<<ROWS, 256, 0, stream>>>(normedb, P[P_LN_G], P[P_LN_B], xlnb, dir);
    launch_gemm(stream, xlnb, D_MOD, 0, P[P_IN_PROJ], 2 * D_INN, 0, 0,
                nullptr, 0, xzb, 2 * D_INN, 0, ROWS, 2 * D_INN, D_MOD, 1, 1.0f, 0);
    xpad_kernel<<<nblk((long long)B_SZ * (S_LEN + 3) * D_INN), 256, 0, stream>>>(xzb, xpadb);
    launch_gemm(stream, xpadb, D_INN, (long long)(S_LEN + 3) * D_INN,
                wconvb, D_INN, 0, 0, nullptr, 0,
                xcb, D_INN, (long long)S_LEN * D_INN,
                S_LEN, D_INN, D_CONV * D_INN, B_SZ, 1.0f, 0);
    bias_silu_kernel<<<nblk((long long)ROWS * D_INN), 256, 0, stream>>>(
        xcb, P[P_CONV_B], xactb);
    launch_gemm(stream, xactb, D_INN, 0, P[P_X_PROJ], 129, 0, 0,
                nullptr, 0, xssmb, 129, 0, ROWS, 129, D_INN, 1, 1.0f, 0);
    delta_kernel<<<nblk((long long)ROWS * D_INN), 256, 0, stream>>>(
        xssmb, P[P_DT_W], P[P_DT_B], deltab);
    scan_kernel<<<dim3(D_INN / 128, B_SZ), 256, 0, stream>>>(
        xactb, deltab, xssmb, causal, charadj,
        P[P_ROUTE_W], P[P_ROUTE_B], histb, yb, dir);
    ygate_kernel<<<nblk((long long)ROWS * D_INN), 256, 0, stream>>>(yb, xzb);
    launch_gemm(stream, yb, D_INN, 0, P[P_MAMBA_OUT], D_MOD, 0, 0,
                nullptr, 0, projs[dir], D_MOD, 0, ROWS, D_MOD, D_INN, 1, 1.0f, 0);
  }
  combine_kernel<<<nblk((long long)ROWS * D_MOD), 256, 0, stream>>>(
      normedb, projb0, projb1, globb);

  // ---------------- 4. residual gate ----------------
  launch_gemm(stream, globb, D_MOD, 0, P[P_RG_W] + (size_t)D_MOD * D_MOD, D_MOD, 0, 0,
              P[P_RG_B], 0, ggb, D_MOD, 0, ROWS, D_MOD, D_MOD, 1, 1.0f, 0);
  launch_gemm(stream, scene, D_MOD, 0, P[P_RG_W], D_MOD, 0, 0,
              nullptr, 0, sg, D_MOD, 0, FLATROWS, D_MOD, D_MOD, 1, 1.0f, 0);
  final_gate_kernel<<<nblk((long long)FLATROWS * D_MOD), 256, 0, stream>>>(
      scene, sg, ggb, globb, out);
}
